// MAGNODecoder_72816875536553
// MI455X (gfx1250) — matrix operand order repaired and checked by reading, not compiled, not run on hardware
//
#include <hip/hip_runtime.h>
#include <math.h>

typedef __attribute__((ext_vector_type(16))) _Float16 v16h;
typedef __attribute__((ext_vector_type(8)))  _Float16 v8h;
typedef __attribute__((ext_vector_type(8)))  float    v8f;

#define N_LAT 4096
#define N_QRY 16384
#define BATCH 4

__device__ __forceinline__ v8f wmma_f16(v16h a, v16h b, v8f c) {
  return __builtin_amdgcn_wmma_f32_16x16x32_f16(false, a, false, b, (short)0, c, false, false);
}

// Branch-free tanh-approx gelu: gelu(x) = x * sigmoid(2*k0*(x + k1*x^3)).
// sigmoid via one v_exp_f32 + one v_rcp_f32 (TRANS ops co-execute with WMMA).
__device__ __forceinline__ float gelu_f(float x) {
  const float k0 = 0.7978845608028654f;       // sqrt(2/pi)
  const float k1 = 0.044715f;
  const float log2e = 1.4426950408889634f;
  float u = 2.0f * k0 * (x + k1 * x * x * x);
  float t = __builtin_amdgcn_exp2f(-log2e * u);   // exp(-u)
  return x * __builtin_amdgcn_rcpf(1.0f + t);     // x * sigmoid(u)
}

// Load a 16x32 f16 A-fragment (K-chunk c) from an LDS row.
// ISA A layout: lane M = lane%16, g = lane/16; element e -> K = (e<8?e:e+8)+8g.
__device__ __forceinline__ v16h load_A(const _Float16* rowbase, int g, int c) {
  const v8h lo = *(const v8h*)(rowbase + c * 32 + 8 * g);        // K = 8g..8g+7
  const v8h hi = *(const v8h*)(rowbase + c * 32 + 16 + 8 * g);   // K = 16+8g..+7
  return __builtin_shufflevector(lo, hi, 0,1,2,3,4,5,6,7,8,9,10,11,12,13,14,15);
}

// ---------------------------------------------------------------------------
__global__ void zero_kernel(float* __restrict__ p, size_t n) {
  size_t i = (size_t)blockIdx.x * blockDim.x + threadIdx.x;
  size_t stride = (size_t)gridDim.x * blockDim.x;
  for (; i < n; i += stride) p[i] = 0.0f;
}

// ---------------------------------------------------------------------------
// Row-major f32 weight (Ksrc x Nsrc) -> f16 WMMA B layout.
// Per (K-chunk c, N-tile t): lane L -> column n=16t+(L&15), element e -> K=32c+(e<8?e:e+8)+8*(L>>4).
__global__ void prep_B(const float* __restrict__ W, _Float16* __restrict__ dst,
                       int Ksrc, int Nsrc, int nChunks, int nTiles) {
  int idx = blockIdx.x * blockDim.x + threadIdx.x;
  int total = nChunks * nTiles * 512;
  if (idx >= total) return;
  int e = idx & 15;
  int L = (idx >> 4) & 31;
  int t = (idx >> 9) % nTiles;
  int c = (idx >> 9) / nTiles;
  int g = L >> 4;
  int n = t * 16 + (L & 15);
  int k = c * 32 + ((e < 8) ? e : e + 8) + 8 * g;
  float v = (k < Ksrc && n < Nsrc) ? W[k * Nsrc + n] : 0.0f;
  dst[idx] = (_Float16)v;
}

// ---------------------------------------------------------------------------
// w = softmax(relu(qc @ Wsw0 + bsw0) @ Wsw1 + bsw1), per query.
__global__ void switch_kernel(const float* __restrict__ qc,
                              const float* __restrict__ Wsw0, const float* __restrict__ bsw0,
                              const float* __restrict__ Wsw1, const float* __restrict__ bsw1,
                              float* __restrict__ w) {
  int q = blockIdx.x * blockDim.x + threadIdx.x;
  if (q >= N_QRY) return;
  const float log2e = 1.4426950408889634f;
  float x0 = qc[q * 2 + 0], x1 = qc[q * 2 + 1];
  float z0 = bsw1[0], z1 = bsw1[1];
#pragma unroll
  for (int j = 0; j < 16; j++) {
    float h = fmaxf(x0 * Wsw0[j] + x1 * Wsw0[16 + j] + bsw0[j], 0.0f);
    z0 += h * Wsw1[j * 2 + 0];
    z1 += h * Wsw1[j * 2 + 1];
  }
  float m = fmaxf(z0, z1);
  float e0 = __builtin_amdgcn_exp2f(log2e * (z0 - m));
  float e1 = __builtin_amdgcn_exp2f(log2e * (z1 - m));
  float inv = __builtin_amdgcn_rcpf(e0 + e1);
  w[q * 2 + 0] = e0 * inv;
  w[q * 2 + 1] = e1 * inv;
}

// ---------------------------------------------------------------------------
// One wave = 16 edges. Kernel MLP 4->64->64->64 via WMMA; segmented scatter
// of k[m,n]*rndata[b,src[m],n] into s[qry[m]][b][n] (qry sorted -> run-length
// reduce in registers, atomic only at segment boundaries).
__global__ void __launch_bounds__(256) edge_kernel(
    const float* __restrict__ latc, const float* __restrict__ qc,
    const float* __restrict__ rnd,
    const int* __restrict__ src, const int* __restrict__ qry,
    const _Float16* __restrict__ B0, const float* __restrict__ b0,
    const _Float16* __restrict__ B1, const float* __restrict__ b1,
    const _Float16* __restrict__ B2, const float* __restrict__ b2,
    float* __restrict__ s, float* __restrict__ cnt) {
  __shared__ __align__(16) _Float16 hbuf[8][16 * 64];   // per-wave 2KB transpose slice
  const int lane = threadIdx.x & 31;
  const int wid  = threadIdx.x >> 5;
  const int base = (blockIdx.x * 8 + wid) * 16;
  const int M = lane & 15, g = lane >> 4;

  const int e0i = base + M;
  const int sM = src[e0i], qM = qry[e0i];
  if (lane < 16) atomicAdd(&cnt[qM], 1.0f);

  // A0 = feat tile (16 x 4) in a K=32 frame; K=0..3 live in g==0, e=0..3.
  v16h a0;
#pragma unroll
  for (int i = 0; i < 16; i++) a0[i] = (_Float16)0.0f;
  if (g == 0) {
    a0[0] = (_Float16)latc[sM * 2 + 0];
    a0[1] = (_Float16)latc[sM * 2 + 1];
    a0[2] = (_Float16)qc[qM * 2 + 0];
    a0[3] = (_Float16)qc[qM * 2 + 1];
  }

  _Float16* hrow = &hbuf[wid][0];

  // ---- layer 1: (16x4) @ Wk0 -> 16x64, gelu, to LDS (C->A transpose)
#pragma unroll
  for (int t = 0; t < 4; t++) {
    v8f acc;
#pragma unroll
    for (int i = 0; i < 8; i++) acc[i] = 0.0f;
    acc = wmma_f16(a0, ((const v16h*)B0)[t * 32 + lane], acc);
    float bias = b0[t * 16 + M];
#pragma unroll
    for (int r = 0; r < 8; r++)
      hrow[(r + 8 * g) * 64 + t * 16 + M] = (_Float16)gelu_f(acc[r] + bias);
  }

  // ---- layer 2: (16x64) @ Wk1 -> 16x64
  v8f acc2[4];
  {
    v16h aL = load_A(hrow + M * 64, g, 0);
    v16h aH = load_A(hrow + M * 64, g, 1);
#pragma unroll
    for (int t = 0; t < 4; t++) {
#pragma unroll
      for (int i = 0; i < 8; i++) acc2[t][i] = 0.0f;
      acc2[t] = wmma_f16(aL, ((const v16h*)B1)[(0 * 4 + t) * 32 + lane], acc2[t]);
      acc2[t] = wmma_f16(aH, ((const v16h*)B1)[(1 * 4 + t) * 32 + lane], acc2[t]);
    }
  }
#pragma unroll
  for (int t = 0; t < 4; t++) {
    float bias = b1[t * 16 + M];
#pragma unroll
    for (int r = 0; r < 8; r++)
      hrow[(r + 8 * g) * 64 + t * 16 + M] = (_Float16)gelu_f(acc2[t][r] + bias);
  }

  // ---- layer 3: (16x64) @ Wk2 -> k (16x64), no activation
  v8f acc3[4];
  {
    v16h aL = load_A(hrow + M * 64, g, 0);
    v16h aH = load_A(hrow + M * 64, g, 1);
#pragma unroll
    for (int t = 0; t < 4; t++) {
#pragma unroll
      for (int i = 0; i < 8; i++) acc3[t][i] = 0.0f;
      acc3[t] = wmma_f16(aL, ((const v16h*)B2)[(0 * 4 + t) * 32 + lane], acc3[t]);
      acc3[t] = wmma_f16(aH, ((const v16h*)B2)[(1 * 4 + t) * 32 + lane], acc3[t]);
    }
  }

  // ---- segmented scatter: s[qry][b][n] += k[m][n] * rndata[b][src][n]
  int sArr[8], qArr[8];
#pragma unroll
  for (int r = 0; r < 8; r++) {
    int ei = base + r + 8 * g;
    sArr[r] = src[ei];
    qArr[r] = qry[ei];
  }
  // Flush predicate: end of this lane-half's 8-row run, or qry changes.
  // Uniform within each 16-lane half (half-wave exec divergence only).
  bool flush[8];
#pragma unroll
  for (int r = 0; r < 8; r++)
    flush[r] = (r == 7) || (qArr[r + 1] != qArr[r]);

#pragma unroll
  for (int t = 0; t < 4; t++) {
    int n = t * 16 + M;
    float bias = b2[n];
#pragma unroll
    for (int bb = 0; bb < 4; bb++) {
      float run = 0.0f;
#pragma unroll
      for (int r = 0; r < 8; r++) {
        float kv = acc3[t][r] + bias;
        float fy = rnd[((size_t)(bb * N_LAT + sArr[r])) * 64 + n];
        run = fmaf(kv, fy, run);
        if (flush[r]) {
          atomicAdd(&s[(size_t)qArr[r] * (BATCH * 64) + bb * 64 + n], run);
          run = 0.0f;
        }
      }
    }
  }
}

// ---------------------------------------------------------------------------
// One wave = 16 rows of decoded (b,q). decoded = w0*mean0 + w1*mean1 (64),
// then gelu(decoded @ Wp0 + bp0) @ Wp1 + bp1 -> out[b][q][0..3].
__global__ void __launch_bounds__(128) final_kernel(
    const float* __restrict__ s0, const float* __restrict__ cnt0,
    const float* __restrict__ s1, const float* __restrict__ cnt1,
    const float* __restrict__ w,
    const _Float16* __restrict__ BP0, const float* __restrict__ bp0,
    const _Float16* __restrict__ BP1, const float* __restrict__ bp1,
    float* __restrict__ out) {
  __shared__ __align__(16) _Float16 hbuf[4][16 * 256];   // per-wave 8KB hidden slice
  const int lane = threadIdx.x & 31, wid = threadIdx.x >> 5;
  const int tile = blockIdx.x * 4 + wid;
  const int R0 = tile * 16;             // rows = b*N_QRY + q
  const int b  = R0 >> 14;              // N_QRY = 16384
  const int q0 = R0 & (N_QRY - 1);
  const int M = lane & 15, g = lane >> 4;
  const int q = q0 + M;

  const float i0 = w[q * 2 + 0] / fmaxf(cnt0[q], 1.0f);
  const float i1 = w[q * 2 + 1] / fmaxf(cnt1[q], 1.0f);
  const float* p0 = s0 + (size_t)q * (BATCH * 64) + b * 64;
  const float* p1 = s1 + (size_t)q * (BATCH * 64) + b * 64;

  // Build decoded row as two A-fragments (K=64).
  v16h aD[2];
#pragma unroll
  for (int c = 0; c < 2; c++) {
#pragma unroll
    for (int e = 0; e < 16; e++) {
      int k = c * 32 + ((e < 8) ? e : e + 8) + 8 * g;
      aD[c][e] = (_Float16)(i0 * p0[k] + i1 * p1[k]);
    }
  }

  _Float16* hrow = &hbuf[wid][0];
  // hidden = gelu(decoded @ Wp0 + bp0): 16 N-tiles x 2 K-chunks
#pragma unroll
  for (int t = 0; t < 16; t++) {
    v8f acc;
#pragma unroll
    for (int i = 0; i < 8; i++) acc[i] = 0.0f;
    acc = wmma_f16(aD[0], ((const v16h*)BP0)[(0 * 16 + t) * 32 + lane], acc);
    acc = wmma_f16(aD[1], ((const v16h*)BP0)[(1 * 16 + t) * 32 + lane], acc);
    float bias = bp0[t * 16 + M];
#pragma unroll
    for (int r = 0; r < 8; r++)
      hrow[(r + 8 * g) * 256 + t * 16 + M] = (_Float16)gelu_f(acc[r] + bias);
  }

  // out = hidden @ Wp1: K=256 in 8 chunks, N=4 (padded to 16)
  v8f accO;
#pragma unroll
  for (int i = 0; i < 8; i++) accO[i] = 0.0f;
#pragma unroll
  for (int c = 0; c < 8; c++) {
    v16h aH = load_A(hrow + M * 256, g, c);
    accO = wmma_f16(aH, ((const v16h*)BP1)[c * 32 + lane], accO);
  }

  if (M < 4) {
    float bias = bp1[M];
#pragma unroll
    for (int r = 0; r < 8; r++) {
      int qq = q0 + r + 8 * g;
      out[((size_t)b * N_QRY + qq) * 4 + M] = accO[r] + bias;
    }
  }
}

// ---------------------------------------------------------------------------
extern "C" void kernel_launch(void* const* d_in, const int* in_sizes, int n_in,
                              void* d_out, int out_size, void* d_ws, size_t ws_size,
                              hipStream_t stream) {
  const float* latc = (const float*)d_in[0];
  const float* rnd  = (const float*)d_in[1];
  const float* qc   = (const float*)d_in[2];
  const int* src0 = (const int*)d_in[3];
  const int* qry0 = (const int*)d_in[4];
  const int* src1 = (const int*)d_in[5];
  const int* qry1 = (const int*)d_in[6];
  const float* Wk0 = (const float*)d_in[7];   const float* bk0 = (const float*)d_in[8];
  const float* Wk1 = (const float*)d_in[9];   const float* bk1 = (const float*)d_in[10];
  const float* Wk2 = (const float*)d_in[11];  const float* bk2 = (const float*)d_in[12];
  const float* Wsw0 = (const float*)d_in[13]; const float* bsw0 = (const float*)d_in[14];
  const float* Wsw1 = (const float*)d_in[15]; const float* bsw1 = (const float*)d_in[16];
  const float* Wp0 = (const float*)d_in[17];  const float* bp0 = (const float*)d_in[18];
  const float* Wp1 = (const float*)d_in[19];  const float* bp1 = (const float*)d_in[20];
  const int E0 = in_sizes[3];
  const int E1 = in_sizes[5];

  // Workspace layout (all offsets 32B-aligned).
  char* ws = (char*)d_ws;
  float* s0   = (float*)ws;                                  // 16 MB
  float* s1   = (float*)(ws + 16777216);                     // 16 MB
  float* cnt0 = (float*)(ws + 2 * 16777216);                 // 64 KB
  float* cnt1 = cnt0 + N_QRY;                                // 64 KB
  float* wsw  = cnt1 + N_QRY;                                // 128 KB (w gates)
  _Float16* B0  = (_Float16*)(wsw + 2 * N_QRY);              // 2048 h
  _Float16* B1  = B0 + 2048;                                 // 4096 h
  _Float16* B2  = B1 + 4096;                                 // 4096 h
  _Float16* BP0 = B2 + 4096;                                 // 16384 h
  _Float16* BP1 = BP0 + 16384;                               // 4096 h

  // 1) clear sums + counts (contiguous region)
  size_t nz = (size_t)2 * N_QRY * BATCH * 64 + 2 * N_QRY;
  zero_kernel<<<2048, 256, 0, stream>>>(s0, nz);

  // 2) swizzle weights into WMMA B layout
  prep_B<<<(512 * 1 * 4 + 255) / 256, 256, 0, stream>>>(Wk0, B0, 4, 64, 1, 4);
  prep_B<<<(512 * 2 * 4 + 255) / 256, 256, 0, stream>>>(Wk1, B1, 64, 64, 2, 4);
  prep_B<<<(512 * 2 * 4 + 255) / 256, 256, 0, stream>>>(Wk2, B2, 64, 64, 2, 4);
  prep_B<<<(512 * 2 * 16 + 255) / 256, 256, 0, stream>>>(Wp0, BP0, 64, 256, 2, 16);
  prep_B<<<(512 * 8 * 1 + 255) / 256, 256, 0, stream>>>(Wp1, BP1, 256, 4, 8, 1);

  // 3) softmax gates
  switch_kernel<<<N_QRY / 256, 256, 0, stream>>>(qc, Wsw0, bsw0, Wsw1, bsw1, wsw);

  // 4) edge MLP + segmented segment-sum scatter (both edge sets)
  edge_kernel<<<E0 / 128, 256, 0, stream>>>(latc, qc, rnd, src0, qry0,
                                            B0, bk0, B1, bk1, B2, bk2, s0, cnt0);
  edge_kernel<<<E1 / 128, 256, 0, stream>>>(latc, qc, rnd, src1, qry1,
                                            B0, bk0, B1, bk1, B2, bk2, s1, cnt1);

  // 5) mean + gate + projection MLP -> out
  final_kernel<<<(BATCH * N_QRY / 16) / 4, 128, 0, stream>>>(
      s0, cnt0, s1, cnt1, wsw, BP0, bp0, BP1, bp1, (float*)d_out);
}
